// HydraAttention_10599979287185
// MI455X (gfx1250) — compile-verified, matrix operand order checked by reading, not run on hardware
//
#include <hip/hip_runtime.h>
#include <stdint.h>

// ---------------------------------------------------------------------------
// Hydra attention, MI455X (gfx1250, wave32, WMMA + async-to-LDS)
//   SEQ=8192, BSZ=4, E=1024, M = SEQ*BSZ = 32768
// ---------------------------------------------------------------------------

typedef __attribute__((ext_vector_type(16))) __bf16 v16bf;
typedef __attribute__((ext_vector_type(8)))  __bf16 v8bf;
typedef __attribute__((ext_vector_type(4)))  __bf16 v4bf;
typedef __attribute__((ext_vector_type(8)))  float  v8f;
typedef __attribute__((ext_vector_type(4)))  float  v4f;

#define SEQ_N  8192
#define BSZ_N  4
#define EDIM   1024
#define MTOT   (SEQ_N * BSZ_N)   // 32768
#define NTOT   (3 * EDIM)        // 3072 fused q|k|v columns

#define BM 128
#define BN 128
#define BK 32
#define LDT 40                   // LDS row pitch in bf16 (80B = 20 banks, conflict-free)
#define NKT (EDIM / BK)          // 32 k-tiles
#define KVCH 32                  // s-chunks for kv reduction

// ---------------------------------------------------------------------------
// CDNA5 async copy: global -> LDS, 16B per lane, tracked by ASYNCcnt.
// GV mode: LDS address from the VDST VGPR, global address from the 64-bit VADDR pair.
// ---------------------------------------------------------------------------
__device__ __forceinline__ void async_ld_b128(void* lds_ptr, const void* gptr) {
  unsigned lds_off = (unsigned)(uintptr_t)lds_ptr;           // low 32b = DS offset
  asm volatile("global_load_async_to_lds_b128 %0, %1, off"
               :: "v"(lds_off), "v"((unsigned long long)(uintptr_t)gptr)
               : "memory");
}
__device__ __forceinline__ void wait_async0() {
  asm volatile("s_wait_asynccnt 0x0" ::: "memory");
}

// ---------------------------------------------------------------------------
// fp32 -> bf16 cast (weights), vectorized x4, n divisible by 4
// ---------------------------------------------------------------------------
__global__ void hydra_cast_bf16(const float* __restrict__ src,
                                __bf16* __restrict__ dst, int n4) {
  int i = blockIdx.x * blockDim.x + threadIdx.x;
  if (i < n4) {
    v4f x = *(const v4f*)(src + (size_t)i * 4);
    v4bf h;
    h[0] = (__bf16)x.x; h[1] = (__bf16)x.y;
    h[2] = (__bf16)x.z; h[3] = (__bf16)x.w;
    *(v4bf*)(dst + (size_t)i * 4) = h;
  }
}

// ---------------------------------------------------------------------------
// LDS fragment load: 16 bf16 per lane.
// v16bf elements [0..7]  = K (koff .. koff+7)
//       elements [8..15] = K (koff+16 .. koff+23), koff = lane<16 ? 0 : 8
// ---------------------------------------------------------------------------
__device__ __forceinline__ v16bf hydra_frag(const __bf16* p) {
  v8bf lo = *(const v8bf*)(p);
  v8bf hi = *(const v8bf*)(p + 16);
  return __builtin_shufflevector(lo, hi, 0, 1, 2, 3, 4, 5, 6, 7,
                                         8, 9, 10, 11, 12, 13, 14, 15);
}

// ---------------------------------------------------------------------------
// GEMM1: fused in-projection.  C[m,n] = sum_k A[m,k] * W[n,k] + bias[n]
//   A = query/key/value (fp32, selected by N-block), W = in_proj bf16 (3072x1024)
//   double-buffered LDS; B tile via global_load_async_to_lds_b128
// ---------------------------------------------------------------------------
__global__ __launch_bounds__(256)
void hydra_gemm_inproj(const float* __restrict__ q_in,
                       const float* __restrict__ k_in,
                       const float* __restrict__ v_in,
                       const __bf16* __restrict__ wb,
                       const float* __restrict__ bias,
                       __bf16* __restrict__ qb,
                       __bf16* __restrict__ kb,
                       __bf16* __restrict__ vb) {
  __shared__ __align__(16) __bf16 lda[2][BM * LDT];
  __shared__ __align__(16) __bf16 ldb[2][BN * LDT];

  const int nb = blockIdx.x;           // 0..23
  const int mb = blockIdx.y;           // 0..255
  const int n0 = nb * BN;
  const int m0 = mb * BM;
  const int sel = n0 / EDIM;           // 0=q 1=k 2=v
  const float* A = (sel == 0) ? q_in : (sel == 1) ? k_in : v_in;
  __bf16* outp   = (sel == 0) ? qb   : (sel == 1) ? kb   : vb;

  const int tid  = threadIdx.x;
  const int lane = tid & 31;
  const int wave = tid >> 5;
  const int wm = wave >> 2;            // 0..1 : 64-row slab
  const int wn = wave & 3;             // 0..3 : 32-col slab
  const int lr = lane & 15;
  const int koff = (lane < 16) ? 0 : 8;

  v8f acc[4][2];
#pragma unroll
  for (int i = 0; i < 4; ++i)
#pragma unroll
    for (int j = 0; j < 2; ++j)
      acc[i][j] = (v8f){0.f, 0.f, 0.f, 0.f, 0.f, 0.f, 0.f, 0.f};

  auto fill = [&](int buf, int kt) {
    const int k0 = kt * BK;
    // A tile: 128x32 fp32 -> bf16 (4 passes of 32 rows)
    const int ra = tid >> 3;           // 0..31
    const int ca = (tid & 7) * 4;      // 0,4,..,28
#pragma unroll
    for (int p = 0; p < 4; ++p) {
      const int row = p * 32 + ra;
      v4f x = *(const v4f*)(A + (size_t)(m0 + row) * EDIM + k0 + ca);
      v4bf h;
      h[0] = (__bf16)x.x; h[1] = (__bf16)x.y;
      h[2] = (__bf16)x.z; h[3] = (__bf16)x.w;
      *(v4bf*)&lda[buf][row * LDT + ca] = h;
    }
    // B tile: 128x32 bf16 via async DMA to LDS (2 passes of 64 rows)
    const int rb = tid >> 2;           // 0..63
    const int cb = (tid & 3) * 8;      // 0,8,16,24
#pragma unroll
    for (int p = 0; p < 2; ++p) {
      const int row = p * 64 + rb;
      async_ld_b128(&ldb[buf][row * LDT + cb],
                    wb + (size_t)(n0 + row) * EDIM + k0 + cb);
    }
  };

  fill(0, 0);
  wait_async0();
  __syncthreads();

  for (int kt = 0; kt < NKT; ++kt) {
    const int cur = kt & 1;
    if (kt + 1 < NKT) fill(cur ^ 1, kt + 1);   // overlap with compute below

    v16bf af[4], bfm[2];
#pragma unroll
    for (int i = 0; i < 4; ++i)
      af[i] = hydra_frag(&lda[cur][(wm * 64 + i * 16 + lr) * LDT + koff]);
#pragma unroll
    for (int j = 0; j < 2; ++j)
      bfm[j] = hydra_frag(&ldb[cur][(wn * 32 + j * 16 + lr) * LDT + koff]);
#pragma unroll
    for (int i = 0; i < 4; ++i)
#pragma unroll
      for (int j = 0; j < 2; ++j)
        acc[i][j] = __builtin_amdgcn_wmma_f32_16x16x32_bf16(
            false, af[i], false, bfm[j], (short)0, acc[i][j], false, false);

    wait_async0();
    __syncthreads();
  }

  // ---- epilogue: + bias, store bf16 ----
  const int rbase = m0 + wm * 64 + ((lane >= 16) ? 8 : 0);
#pragma unroll
  for (int j = 0; j < 2; ++j) {
    const int nglob = n0 + wn * 32 + j * 16 + lr;
    const float bv = bias[nglob];
    const int ncol = nglob - sel * EDIM;
#pragma unroll
    for (int i = 0; i < 4; ++i)
#pragma unroll
      for (int e = 0; e < 8; ++e) {
        const int r = rbase + i * 16 + e;
        outp[(size_t)r * EDIM + ncol] = (__bf16)(acc[i][j][e] + bv);
      }
  }
}

// ---------------------------------------------------------------------------
// Per-row 1/||x|| for q and k (deterministic tree reduction)
// ---------------------------------------------------------------------------
__global__ __launch_bounds__(256)
void hydra_row_stats(const __bf16* __restrict__ qb,
                     const __bf16* __restrict__ kb,
                     float* __restrict__ qrs, float* __restrict__ krs) {
  __shared__ float red[256];
  const int m = blockIdx.x;
  const int tid = threadIdx.x;
  float sq = 0.f, sk = 0.f;
  for (int e = tid; e < EDIM; e += 256) {
    const float a = (float)qb[(size_t)m * EDIM + e];
    const float c = (float)kb[(size_t)m * EDIM + e];
    sq += a * a;
    sk += c * c;
  }
  red[tid] = sq; __syncthreads();
  for (int st = 128; st > 0; st >>= 1) {
    if (tid < st) red[tid] += red[tid + st];
    __syncthreads();
  }
  if (tid == 0) qrs[m] = rsqrtf(red[0]);
  __syncthreads();
  red[tid] = sk; __syncthreads();
  for (int st = 128; st > 0; st >>= 1) {
    if (tid < st) red[tid] += red[tid + st];
    __syncthreads();
  }
  if (tid == 0) krs[m] = rsqrtf(red[0]);
}

// ---------------------------------------------------------------------------
// kv[b,e] = sum_s norm_k[s,b,e] * v[s,b,e]  — chunked (deterministic)
// ---------------------------------------------------------------------------
__global__ __launch_bounds__(256)
void hydra_kv_partial(const __bf16* __restrict__ kb,
                      const __bf16* __restrict__ vb,
                      const float* __restrict__ krs,
                      float* __restrict__ part) {
  const int e  = blockIdx.x * 256 + threadIdx.x;  // gridDim.x = 4
  const int b  = blockIdx.y;                      // 4
  const int ch = blockIdx.z;                      // KVCH
  const int s0 = ch * (SEQ_N / KVCH);
  float acc = 0.f;
  for (int s = s0; s < s0 + SEQ_N / KVCH; ++s) {
    const int m = s * BSZ_N + b;
    const size_t idx = (size_t)m * EDIM + e;
    acc += (float)kb[idx] * krs[m] * (float)vb[idx];
  }
  part[((size_t)ch * BSZ_N + b) * EDIM + e] = acc;
}

__global__ __launch_bounds__(256)
void hydra_kv_reduce(const float* __restrict__ part, float* __restrict__ kv) {
  const int e = blockIdx.x * 256 + threadIdx.x;
  const int b = blockIdx.y;
  float acc = 0.f;
  for (int c = 0; c < KVCH; ++c)
    acc += part[((size_t)c * BSZ_N + b) * EDIM + e];
  kv[(size_t)b * EDIM + e] = acc;
}

// ---------------------------------------------------------------------------
// GEMM2: out-projection.  A[m,e] = bf16( q_bf16[m,e] * qrs[m] * kv[b(m),e] )
//   C[m,f] = sum_e A[m,e] * Wout[f,e] + bias_out[f]   (fp32 output)
//   double-buffered LDS; B tile via global_load_async_to_lds_b128
// ---------------------------------------------------------------------------
__global__ __launch_bounds__(256)
void hydra_gemm_outproj(const __bf16* __restrict__ qb,
                        const float* __restrict__ qrs,
                        const float* __restrict__ kv,
                        const __bf16* __restrict__ wb,
                        const float* __restrict__ bias,
                        float* __restrict__ out) {
  __shared__ __align__(16) __bf16 lda[2][BM * LDT];
  __shared__ __align__(16) __bf16 ldb[2][BN * LDT];

  const int nb = blockIdx.x;           // 0..7
  const int mb = blockIdx.y;           // 0..255
  const int n0 = nb * BN;
  const int m0 = mb * BM;

  const int tid  = threadIdx.x;
  const int lane = tid & 31;
  const int wave = tid >> 5;
  const int wm = wave >> 2;
  const int wn = wave & 3;
  const int lr = lane & 15;
  const int koff = (lane < 16) ? 0 : 8;

  v8f acc[4][2];
#pragma unroll
  for (int i = 0; i < 4; ++i)
#pragma unroll
    for (int j = 0; j < 2; ++j)
      acc[i][j] = (v8f){0.f, 0.f, 0.f, 0.f, 0.f, 0.f, 0.f, 0.f};

  auto fill = [&](int buf, int kt) {
    const int k0 = kt * BK;
    const int r = tid >> 2;            // 0..63
    const int c = (tid & 3) * 8;       // 0,8,16,24
    // A tile: on-the-fly normalize * kv broadcast -> bf16 LDS (2 passes)
#pragma unroll
    for (int p = 0; p < 2; ++p) {
      const int row = p * 64 + r;
      const int m = m0 + row;
      v8bf x = *(const v8bf*)(qb + (size_t)m * EDIM + k0 + c);
      const float s = qrs[m];
      const float* kvp = kv + (size_t)(m & (BSZ_N - 1)) * EDIM + k0 + c;
      v8bf h;
#pragma unroll
      for (int u = 0; u < 8; ++u)
        h[u] = (__bf16)((float)x[u] * s * kvp[u]);
      *(v8bf*)&lda[buf][row * LDT + c] = h;
    }
    // B tile: out_proj weights via async DMA to LDS (2 passes)
#pragma unroll
    for (int p = 0; p < 2; ++p) {
      const int row = p * 64 + r;
      async_ld_b128(&ldb[buf][row * LDT + c],
                    wb + (size_t)(n0 + row) * EDIM + k0 + c);
    }
  };

  fill(0, 0);
  wait_async0();
  __syncthreads();

  for (int kt = 0; kt < NKT; ++kt) {
    const int cur = kt & 1;
    if (kt + 1 < NKT) fill(cur ^ 1, kt + 1);

    v16bf af[4], bfm[2];
#pragma unroll
    for (int i = 0; i < 4; ++i)
      af[i] = hydra_frag(&lda[cur][(wm * 64 + i * 16 + lr) * LDT + koff]);
#pragma unroll
    for (int j = 0; j < 2; ++j)
      bfm[j] = hydra_frag(&ldb[cur][(wn * 32 + j * 16 + lr) * LDT + koff]);
#pragma unroll
    for (int i = 0; i < 4; ++i)
#pragma unroll
      for (int j = 0; j < 2; ++j)
        acc[i][j] = __builtin_amdgcn_wmma_f32_16x16x32_bf16(
            false, af[i], false, bfm[j], (short)0, acc[i][j], false, false);

    wait_async0();
    __syncthreads();
  }

  const int rbase = m0 + wm * 64 + ((lane >= 16) ? 8 : 0);
#pragma unroll
  for (int j = 0; j < 2; ++j) {
    const int nglob = n0 + wn * 32 + j * 16 + lr;
    const float bv = bias[nglob];
#pragma unroll
    for (int i = 0; i < 4; ++i)
#pragma unroll
      for (int e = 0; e < 8; ++e) {
        const int r = rbase + i * 16 + e;
        out[(size_t)r * EDIM + nglob] = acc[i][j][e] + bv;
      }
  }
}

// ---------------------------------------------------------------------------
// Workspace layout (all sizes 256B-aligned)
// ---------------------------------------------------------------------------
static constexpr size_t SZ_ACT  = (size_t)MTOT * EDIM * 2;   // 64 MiB each
static constexpr size_t OFF_QB  = 0;
static constexpr size_t OFF_KB  = OFF_QB + SZ_ACT;
static constexpr size_t OFF_VB  = OFF_KB + SZ_ACT;
static constexpr size_t OFF_WIN = OFF_VB + SZ_ACT;           // 3072x1024 bf16
static constexpr size_t SZ_WIN  = (size_t)NTOT * EDIM * 2;
static constexpr size_t OFF_WOUT = OFF_WIN + SZ_WIN;         // 1024x1024 bf16
static constexpr size_t SZ_WOUT = (size_t)EDIM * EDIM * 2;
static constexpr size_t OFF_QRS = OFF_WOUT + SZ_WOUT;        // 32768 f32
static constexpr size_t SZ_RS   = (size_t)MTOT * 4;
static constexpr size_t OFF_KRS = OFF_QRS + SZ_RS;
static constexpr size_t OFF_PART = OFF_KRS + SZ_RS;          // 32x4x1024 f32
static constexpr size_t SZ_PART = (size_t)KVCH * BSZ_N * EDIM * 4;
static constexpr size_t OFF_KV  = OFF_PART + SZ_PART;        // 4x1024 f32

extern "C" void kernel_launch(void* const* d_in, const int* in_sizes, int n_in,
                              void* d_out, int out_size, void* d_ws, size_t ws_size,
                              hipStream_t stream) {
  const float* query = (const float*)d_in[0];
  const float* key_  = (const float*)d_in[1];
  const float* value = (const float*)d_in[2];
  const float* w_in  = (const float*)d_in[3];
  const float* b_in  = (const float*)d_in[4];
  const float* w_out = (const float*)d_in[5];
  const float* b_out = (const float*)d_in[6];
  float* out = (float*)d_out;

  char* ws = (char*)d_ws;
  __bf16* qb     = (__bf16*)(ws + OFF_QB);
  __bf16* kb     = (__bf16*)(ws + OFF_KB);
  __bf16* vb     = (__bf16*)(ws + OFF_VB);
  __bf16* win_b  = (__bf16*)(ws + OFF_WIN);
  __bf16* wout_b = (__bf16*)(ws + OFF_WOUT);
  float*  qrs    = (float*)(ws + OFF_QRS);
  float*  krs    = (float*)(ws + OFF_KRS);
  float*  part   = (float*)(ws + OFF_PART);
  float*  kv     = (float*)(ws + OFF_KV);

  // 1) weights -> bf16
  {
    const int n4a = NTOT * EDIM / 4;        // 786432
    const int n4b = EDIM * EDIM / 4;        // 262144
    hydra_cast_bf16<<<(n4a + 255) / 256, 256, 0, stream>>>(w_in, win_b, n4a);
    hydra_cast_bf16<<<(n4b + 255) / 256, 256, 0, stream>>>(w_out, wout_b, n4b);
  }
  // 2) fused in-projection (q|k|v), bias added, bf16 outputs
  hydra_gemm_inproj<<<dim3(NTOT / BN, MTOT / BM), 256, 0, stream>>>(
      query, key_, value, win_b, b_in, qb, kb, vb);
  // 3) per-row reciprocal norms for q and k
  hydra_row_stats<<<MTOT, 256, 0, stream>>>(qb, kb, qrs, krs);
  // 4) kv = sum_s norm(k) * v   (two-stage, deterministic)
  hydra_kv_partial<<<dim3(EDIM / 256, BSZ_N, KVCH), 256, 0, stream>>>(kb, vb, krs, part);
  hydra_kv_reduce<<<dim3(EDIM / 256, BSZ_N), 256, 0, stream>>>(part, kv);
  // 5) out-projection on (norm(q) * kv), fp32 output + bias
  hydra_gemm_outproj<<<dim3(EDIM / BN, MTOT / BM), 256, 0, stream>>>(
      qb, qrs, kv, wout_b, b_out, out);
}
